// PoissonGaussianReadout_45131516346520
// MI455X (gfx1250) — matrix operand order repaired and checked by reading, not compiled, not run on hardware
//
#include <hip/hip_runtime.h>
#include <hip/hip_bf16.h>

typedef __attribute__((ext_vector_type(16))) _Float16 v16h;
typedef __attribute__((ext_vector_type(8)))  _Float16 v8h;
typedef __attribute__((ext_vector_type(4)))  _Float16 v4h;
typedef __attribute__((ext_vector_type(8)))  float    v8f;
typedef __attribute__((ext_vector_type(4)))  float    v4f;

#define BATCH 16
#define CD 1024      // C*T
#define HW 1296      // 36*36
#define NNEUR 4096
#define SPAD 1040    // padded half-stride for staged samples

// ---------------------------------------------------------------------------
// Kernel 1: transpose x[b][d][s] -> feat[b][s][d]  (makes channel dim contiguous)
// x: [B, C, T, Wd, Hd] = [b][d=c*16+t][s=hin*36+win]
// ---------------------------------------------------------------------------
__global__ __launch_bounds__(256) void k_transpose(const float* __restrict__ x,
                                                   float* __restrict__ feat) {
    __shared__ float tile[32][33];
    const int b  = blockIdx.z;
    const int d0 = blockIdx.y * 32;   // 1024/32 = 32 tiles, always in range
    const int s0 = blockIdx.x * 32;   // 41 tiles, guard s < 1296
    const int tx = threadIdx.x, ty = threadIdx.y;

    const float* xb = x + (size_t)b * CD * HW;
#pragma unroll
    for (int i = 0; i < 4; ++i) {
        int d = d0 + ty + 8 * i;
        int s = s0 + tx;
        if (s < HW) tile[ty + 8 * i][tx] = xb[(size_t)d * HW + s];
    }
    __syncthreads();
    float* fb = feat + (size_t)b * HW * CD;
#pragma unroll
    for (int i = 0; i < 4; ++i) {
        int s = s0 + ty + 8 * i;
        int d = d0 + tx;
        if (s < HW) fb[(size_t)s * CD + d] = tile[tx][ty + 8 * i];
    }
}

// ---------------------------------------------------------------------------
// Kernel 2: one block per neuron. Stage bilinear samples (16 batches x 1024 ch)
// + W row into LDS as f16, then 16 batch dot-products via WMMA f32_16x16x32_f16
// with W replicated across all 16 B-columns (GEMV-by-WMMA). Split-K over 8
// waves, LDS reduce, elu(y)+1.
// ---------------------------------------------------------------------------
__global__ __launch_bounds__(256) void k_readout(const float* __restrict__ feat,
                                                 const float* __restrict__ mu,
                                                 const float* __restrict__ Wm,
                                                 const float* __restrict__ bias,
                                                 float* __restrict__ out) {
    __shared__ alignas(16) _Float16 sh_s[BATCH * SPAD];  // staged samples (f16)
    __shared__ alignas(32) _Float16 sh_w[CD];            // W row (f16)
    __shared__ float red[8][256];                        // cross-wave reduce

    const int n    = blockIdx.x;
    const int tid  = threadIdx.x;
    const int lane = tid & 31;
    const int wv   = tid >> 5;

    // ---- per-neuron bilinear setup (uniform across block) ----
    float gx = fminf(fmaxf(mu[2 * n + 0], -1.f), 1.f);  // indexes win (W_in)
    float gy = fminf(fmaxf(mu[2 * n + 1], -1.f), 1.f);  // indexes hin (H_in)
    float ix = (gx + 1.f) * 18.f - 0.5f;
    float iy = (gy + 1.f) * 18.f - 0.5f;
    float x0f = floorf(ix), y0f = floorf(iy);
    int x0 = (int)x0f, y0 = (int)y0f, x1 = x0 + 1, y1 = y0 + 1;
    float wx1 = ix - x0f, wx0 = 1.f - wx1;
    float wy1 = iy - y0f, wy0 = 1.f - wy1;
    bool vx0 = (x0 >= 0) && (x0 <= 35), vx1 = (x1 >= 0) && (x1 <= 35);
    bool vy0 = (y0 >= 0) && (y0 <= 35), vy1 = (y1 >= 0) && (y1 <= 35);
    float w00 = wy0 * wx0 * ((vy0 && vx0) ? 1.f : 0.f);
    float w01 = wy0 * wx1 * ((vy0 && vx1) ? 1.f : 0.f);
    float w10 = wy1 * wx0 * ((vy1 && vx0) ? 1.f : 0.f);
    float w11 = wy1 * wx1 * ((vy1 && vx1) ? 1.f : 0.f);
    int xc0 = x0 < 0 ? 0 : (x0 > 35 ? 35 : x0);
    int xc1 = x1 < 0 ? 0 : (x1 > 35 ? 35 : x1);
    int yc0 = y0 < 0 ? 0 : (y0 > 35 ? 35 : y0);
    int yc1 = y1 < 0 ? 0 : (y1 > 35 ? 35 : y1);
    int loc00 = yc0 * 36 + xc0, loc01 = yc0 * 36 + xc1;
    int loc10 = yc1 * 36 + xc0, loc11 = yc1 * 36 + xc1;

    // ---- stage W row to LDS (f16): 256 threads x 4 ch ----
    {
        const float* wr = Wm + (size_t)n * CD;
        int k = tid * 4;
        v4f w4 = *(const v4f*)(wr + k);
        v4h h;
        h[0] = (_Float16)w4[0]; h[1] = (_Float16)w4[1];
        h[2] = (_Float16)w4[2]; h[3] = (_Float16)w4[3];
        *(v4h*)(&sh_w[k]) = h;
    }

    // ---- stage bilinear samples: wave wv handles batches 2wv, 2wv+1 ----
#pragma unroll
    for (int bb = 0; bb < 2; ++bb) {
        int b = wv * 2 + bb;
        const float* f00 = feat + ((size_t)b * HW + loc00) * CD;
        const float* f01 = feat + ((size_t)b * HW + loc01) * CD;
        const float* f10 = feat + ((size_t)b * HW + loc10) * CD;
        const float* f11 = feat + ((size_t)b * HW + loc11) * CD;
        _Float16* srow = &sh_s[b * SPAD];
#pragma unroll
        for (int j = 0; j < 8; ++j) {
            int d = j * 128 + lane * 4;  // coalesced float4 per lane
            v4f a0 = *(const v4f*)(f00 + d);
            v4f a1 = *(const v4f*)(f01 + d);
            v4f a2 = *(const v4f*)(f10 + d);
            v4f a3 = *(const v4f*)(f11 + d);
            v4h h;
#pragma unroll
            for (int e = 0; e < 4; ++e) {
                float s = w00 * a0[e] + w01 * a1[e] + w10 * a2[e] + w11 * a3[e];
                h[e] = (_Float16)s;
            }
            *(v4h*)(srow + d) = h;
        }
    }
    __syncthreads();

    // ---- WMMA: D[b, n'] += A[b, k] * Brep[k, n'], all n' columns identical ----
    // A as loaded here (contiguous k order, not the ISA table order): lane m=lane&15, half hi=lane>>4:
    //   elems 0..15 = K [kb + 16*hi .. +15]
    // B (same contiguous k order as A): lane n'(=lane&15): elems = K [kb + 16*hi .. +15]
    v8f acc = {};
    const int m  = lane & 15;
    const int hi = lane >> 4;
    const _Float16* srow = &sh_s[m * SPAD];
#pragma unroll
    for (int c = 0; c < 4; ++c) {
        int kb = wv * 128 + c * 32;  // split-K: wave wv owns K range [128wv, 128wv+128)
        v8h alo = *(const v8h*)(srow + kb + 16 * hi);
        v8h ahi = *(const v8h*)(srow + kb + 16 * hi + 8);
        v16h a = __builtin_shufflevector(alo, ahi, 0, 1, 2, 3, 4, 5, 6, 7,
                                         8, 9, 10, 11, 12, 13, 14, 15);
        v16h bm = *(const v16h*)(&sh_w[kb + 16 * hi]);
        acc = __builtin_amdgcn_wmma_f32_16x16x32_f16(
            /*neg_a=*/false, a, /*neg_b=*/false, bm,
            /*c_mod=*/(short)0, acc, /*reuse_a=*/false, /*reuse_b=*/false);
    }

    // ---- cross-wave split-K reduction ----
#pragma unroll
    for (int i = 0; i < 8; ++i) red[wv][i * 32 + lane] = acc[i];
    __syncthreads();

    // Column n'==0 threads (tid%16==0) own output element (vgpr i=tid/32, lane l=tid%32):
    //   batch b = i + 8*(l>=16)
    if ((tid & 15) == 0) {
        float v = 0.f;
#pragma unroll
        for (int w = 0; w < 8; ++w) v += red[w][tid];
        int i = tid >> 5;
        int l = tid & 31;
        int b = i + 8 * (l >> 4);
        float y = v + bias[n];
        out[(size_t)b * NNEUR + n] = (y > 0.f) ? (y + 1.f) : __expf(y);
    }
}

// ---------------------------------------------------------------------------
extern "C" void kernel_launch(void* const* d_in, const int* in_sizes, int n_in,
                              void* d_out, int out_size, void* d_ws, size_t ws_size,
                              hipStream_t stream) {
    const float* x    = (const float*)d_in[0];  // [16,64,16,36,36]
    const float* mu   = (const float*)d_in[1];  // [4096,2]
    // d_in[2] = sigma (unused in eval mode)
    const float* Wm   = (const float*)d_in[3];  // [4096,1024]
    const float* bias = (const float*)d_in[4];  // [4096]
    float* out = (float*)d_out;                 // [16,4096]

    float* feat = (float*)d_ws;  // [16,1296,1024] f32 = ~85 MB scratch

    k_transpose<<<dim3(41, 32, 16), dim3(32, 8, 1), 0, stream>>>(x, feat);
    k_readout<<<dim3(NNEUR), dim3(256), 0, stream>>>(feat, mu, Wm, bias, out);
}
